// DecoderLayer_58222576665006
// MI455X (gfx1250) — matrix operand order repaired and checked by reading, not compiled, not run on hardware
//
#include <hip/hip_runtime.h>
#include <hip/hip_bf16.h>

// ---------------------------------------------------------------------------
// Decoder layer for MI455X (gfx1250, wave32, WMMA).
//
// scores einsum 'bink,bijm->binm' is an OUTER PRODUCT of row-sums, so QK^T
// collapses to chunk sums. All real GEMMs (projections, AV, FFN) run on
// v_wmma_f32_16x16x32_bf16 with fp32 accumulation; softmax/LN stay fp32.
// ---------------------------------------------------------------------------

typedef __attribute__((ext_vector_type(16))) __bf16 bf16x16;
typedef __attribute__((ext_vector_type(8)))  __bf16 bf16x8;
typedef __attribute__((ext_vector_type(8)))  float  f32x8;
typedef __attribute__((ext_vector_type(4)))  float  f32x4;

#define S_  512
#define B_  8
#define D_  1024
#define H_  16
#define FF_ 4096
#define MD_ 4096   // S_*B_  (rows of all activation GEMMs)

// ---------------------------------------------------------------------------
// fp32 -> bf16 conversion
// ---------------------------------------------------------------------------
__global__ __launch_bounds__(256) void f32_to_bf16_k(const float* __restrict__ in,
                                                     __bf16* __restrict__ out, int n) {
    int i = blockIdx.x * 256 + threadIdx.x;
    if (i < n) out[i] = (__bf16)in[i];
}

// ---------------------------------------------------------------------------
// Tiled bf16 WMMA GEMM:  C[M,N] = A[M,K] * B[K,N]   (A,B bf16 row-major)
//   EPI 0: f32 store   1: f32+bias   2: bf16 relu(x+bias)   3: bf16 store
// Batched via blockIdx.z; B batch offset = (z/bInner)*bsBo + (z%bInner)*bsBi
// (needed for V, whose head slices live at b*S*D + h*64).
// BM=128 fixed; BN in {128,64}. 256 threads = 8 waves; wave tile 32 x (BN/2).
// ---------------------------------------------------------------------------
template <int BM, int BN, int EPI>
__global__ __launch_bounds__(256) void gemm_bf16_wmma(
    const __bf16* __restrict__ A, int lda, long bsA,
    const __bf16* __restrict__ B, int ldb, long bsBo, long bsBi, int bInner,
    void* __restrict__ Cp, int ldc, long bsC,
    const float* __restrict__ bias, int K) {

    constexpr int BK  = 64;
    constexpr int WN  = BN / 2;      // wave tile cols (2 waves across N)
    constexpr int FN  = WN / 16;     // N fragments per wave
    constexpr int PAD = 8;

    __shared__ __bf16 As[BM][BK + PAD];
    __shared__ __bf16 Bs[BN][BK + PAD];   // stored transposed: Bs[n][k]

    const int tid  = threadIdx.x;
    const int wave = tid >> 5;
    const int lane = tid & 31;
    const int wm   = wave & 3;       // 4 waves along M (4*32 = 128)
    const int wn   = wave >> 2;      // 2 waves along N
    const int r16  = lane & 15;
    const int half = lane >> 4;

    const __bf16* Ab = A + (size_t)blockIdx.z * bsA + (size_t)blockIdx.y * BM * lda;
    const __bf16* Bb = B + (size_t)(blockIdx.z / bInner) * bsBo
                         + (size_t)(blockIdx.z % bInner) * bsBi
                         + (size_t)blockIdx.x * BN;

    f32x8 acc[2][FN];
#pragma unroll
    for (int i = 0; i < 2; ++i)
#pragma unroll
        for (int j = 0; j < FN; ++j)
#pragma unroll
            for (int e = 0; e < 8; ++e) acc[i][j][e] = 0.f;

    for (int k0 = 0; k0 < K; k0 += BK) {
        __syncthreads();
        // ---- A tile: BM x 64 bf16, 16B chunks -------------------------------
#pragma unroll 2
        for (int i = tid; i < BM * 8; i += 256) {
            int r = i >> 3, c = (i & 7) * 8;
            *(bf16x8*)&As[r][c] = *(const bf16x8*)(Ab + (size_t)r * lda + k0 + c);
            if (k0 + BK < K)
                __builtin_prefetch(Ab + (size_t)r * lda + k0 + BK + c, 0, 1);
        }
        // ---- B tile: 64 x BN, transposed into LDS ---------------------------
#pragma unroll 2
        for (int i = tid; i < (BN / 8) * 64; i += 256) {
            int kr = i & 63, c = (i >> 6) * 8;
            bf16x8 v = *(const bf16x8*)(Bb + (size_t)(k0 + kr) * ldb + c);
#pragma unroll
            for (int j = 0; j < 8; ++j) Bs[c + j][kr] = v[j];
        }
        __syncthreads();

#pragma unroll
        for (int ks = 0; ks < BK; ks += 32) {
            // A fragment: lane holds row (wm*32+fm*16+r16), K = {8h..8h+7, 16+8h..}
            bf16x16 af[2];
#pragma unroll
            for (int fm = 0; fm < 2; ++fm) {
                const __bf16* p = &As[wm * 32 + fm * 16 + r16][ks + 8 * half];
                bf16x8 lo = *(const bf16x8*)p;
                bf16x8 hi = *(const bf16x8*)(p + 16);
#pragma unroll
                for (int j = 0; j < 8; ++j) { af[fm][j] = lo[j]; af[fm][j + 8] = hi[j]; }
            }
            // B fragment: lane holds col, K = {8h..8h+7, 16+8h..}
            bf16x16 bfm[FN];
#pragma unroll
            for (int fn = 0; fn < FN; ++fn) {
                const __bf16* p = &Bs[wn * WN + fn * 16 + r16][ks + 8 * half];
                bf16x8 lo = *(const bf16x8*)p;
                bf16x8 hi = *(const bf16x8*)(p + 16);
#pragma unroll
                for (int j = 0; j < 8; ++j) { bfm[fn][j] = lo[j]; bfm[fn][j + 8] = hi[j]; }
            }
#pragma unroll
            for (int fm = 0; fm < 2; ++fm)
#pragma unroll
                for (int fn = 0; fn < FN; ++fn)
                    acc[fm][fn] = __builtin_amdgcn_wmma_f32_16x16x32_bf16(
                        false, af[fm], false, bfm[fn],
                        (short)0, acc[fm][fn], false, false);
        }
    }

    // ---- epilogue: C element e -> row tile+e+8*half, col tile+(lane&15) ----
    const size_t cBase   = (size_t)blockIdx.z * bsC;
    const int    colBase = blockIdx.x * BN + wn * WN;
    const int    rowBase = blockIdx.y * BM + wm * 32;
#pragma unroll
    for (int fm = 0; fm < 2; ++fm)
#pragma unroll
        for (int fn = 0; fn < FN; ++fn) {
            int col  = colBase + fn * 16 + r16;
            int row0 = rowBase + fm * 16 + 8 * half;
            float bv = (EPI == 1 || EPI == 2) ? bias[col] : 0.f;
#pragma unroll
            for (int e = 0; e < 8; ++e) {
                float v = acc[fm][fn][e] + bv;
                if (EPI == 2) v = fmaxf(v, 0.f);
                size_t idx = cBase + (size_t)(row0 + e) * ldc + col;
                if (EPI <= 1) ((float*)Cp)[idx]  = v;
                else          ((__bf16*)Cp)[idx] = (__bf16)v;
            }
        }
}

// ---------------------------------------------------------------------------
// Sum of each contiguous 64-float chunk (the per-(row, head) depth sums).
// 16 lanes cooperate per chunk (float4 each), reduce via shuffles.
// ---------------------------------------------------------------------------
__global__ __launch_bounds__(256) void chunk_sum64_k(const float* __restrict__ in,
                                                     float* __restrict__ out, float scale) {
    int t = blockIdx.x * 256 + threadIdx.x;
    int chunk = t >> 4, l = t & 15;
    f32x4 v = *(const f32x4*)(in + ((size_t)chunk << 6) + (l << 2));
    float s = v[0] + v[1] + v[2] + v[3];
    s += __shfl_xor(s, 1, 32);
    s += __shfl_xor(s, 2, 32);
    s += __shfl_xor(s, 4, 32);
    s += __shfl_xor(s, 8, 32);
    if (l == 0) out[chunk] = s * scale;
}

// ---------------------------------------------------------------------------
// Column sums of skew(rel):  out[bh,m] = sum_r skew(rel)[bh,r,m]
// skew(rel)[r,m] = rel[p>>10, p&1023] (0 if col>=512), p = r*1023 + 511 + m
// ---------------------------------------------------------------------------
__global__ __launch_bounds__(256) void skew_colsum_k(const float* __restrict__ rel,
                                                     float* __restrict__ out) {
    int bh = blockIdx.y;
    int m  = blockIdx.x * 256 + threadIdx.x;          // 0..511
    const float* r = rel + (size_t)bh * S_ * S_;
    float s = 0.f;
    for (int ri = 0; ri < S_; ++ri) {
        int p  = ri * (2 * S_ - 1) + (S_ - 1) + m;
        int rr = p >> 10, cc = p & 1023;
        if (cc < S_) s += r[(size_t)rr * S_ + cc];
    }
    out[(size_t)bh * S_ + m] = s;
}

// ---------------------------------------------------------------------------
// Rank-1 softmax: w[bh,n,m] = softmax_m( Qs[n] * (Ks[m]+As[m]) + mask )
// Qs/Ks live in chunk-sum layout: idx = (b*512 + pos)*16 + h. Output bf16.
// ---------------------------------------------------------------------------
__global__ __launch_bounds__(256) void softmax_rank1_k(
    const float* __restrict__ Qs, const float* __restrict__ Ks,
    const float* __restrict__ As, __bf16* __restrict__ W, int causal) {
    __shared__ float red[8];
    int n = blockIdx.x, bh = blockIdx.y;
    int b = bh >> 4, h = bh & 15;
    int tid = threadIdx.x;

    float qv = Qs[((size_t)(b * 512 + n) << 4) + h];
    float sc[2];
#pragma unroll
    for (int i = 0; i < 2; ++i) {
        int m = tid + (i << 8);
        float t = Ks[((size_t)(b * 512 + m) << 4) + h] + As[((size_t)bh << 9) + m];
        float s = qv * t;
        if (causal && m > n) s += -1e9f;
        sc[i] = s;
    }
    float mx = fmaxf(sc[0], sc[1]);
    for (int m = 16; m; m >>= 1) mx = fmaxf(mx, __shfl_xor(mx, m, 32));
    if ((tid & 31) == 0) red[tid >> 5] = mx;
    __syncthreads();
    float bm = red[0];
#pragma unroll
    for (int i = 1; i < 8; ++i) bm = fmaxf(bm, red[i]);
    __syncthreads();

    float e0 = __expf(sc[0] - bm), e1 = __expf(sc[1] - bm);
    float s = e0 + e1;
    for (int m = 16; m; m >>= 1) s += __shfl_xor(s, m, 32);
    if ((tid & 31) == 0) red[tid >> 5] = s;
    __syncthreads();
    float tot = 0.f;
#pragma unroll
    for (int i = 0; i < 8; ++i) tot += red[i];
    float inv = 1.f / tot;

    size_t wb = ((size_t)bh * 512 + n) * 512;
    W[wb + tid]       = (__bf16)(e0 * inv);
    W[wb + tid + 256] = (__bf16)(e1 * inv);
}

// ---------------------------------------------------------------------------
// Fused residual + LayerNorm over D=1024; optional bf16 copy for next GEMM.
// One 256-thread block per row; float4 per thread.
// ---------------------------------------------------------------------------
__global__ __launch_bounds__(256) void residual_ln_k(
    const float* __restrict__ a, const float* __restrict__ b,
    const float* __restrict__ g, const float* __restrict__ bet,
    float* __restrict__ outF, __bf16* __restrict__ outB) {
    __shared__ float s1[8], s2[8];
    int row = blockIdx.x, tid = threadIdx.x;
    size_t base = (size_t)row * D_ + tid * 4;
    f32x4 va = *(const f32x4*)(a + base);
    f32x4 vb = *(const f32x4*)(b + base);
    f32x4 v;
#pragma unroll
    for (int j = 0; j < 4; ++j) v[j] = va[j] + vb[j];

    float s = v[0] + v[1] + v[2] + v[3];
    float q = v[0]*v[0] + v[1]*v[1] + v[2]*v[2] + v[3]*v[3];
    for (int m = 16; m; m >>= 1) { s += __shfl_xor(s, m, 32); q += __shfl_xor(q, m, 32); }
    if ((tid & 31) == 0) { s1[tid >> 5] = s; s2[tid >> 5] = q; }
    __syncthreads();
    float Ssum = 0.f, Qsum = 0.f;
#pragma unroll
    for (int i = 0; i < 8; ++i) { Ssum += s1[i]; Qsum += s2[i]; }
    float mu  = Ssum * (1.f / D_);
    float var = Qsum * (1.f / D_) - mu * mu;
    float rs  = rsqrtf(var + 1e-5f);
#pragma unroll
    for (int j = 0; j < 4; ++j) {
        int c = tid * 4 + j;
        float y = (v[j] - mu) * rs * g[c] + bet[c];
        outF[base + j] = y;
        if (outB) outB[base + j] = (__bf16)y;
    }
}

// ---------------------------------------------------------------------------
// Launcher
// ---------------------------------------------------------------------------
extern "C" void kernel_launch(void* const* d_in, const int* in_sizes, int n_in,
                              void* d_out, int out_size, void* d_ws, size_t ws_size,
                              hipStream_t stream) {
    (void)in_sizes; (void)n_in; (void)out_size; (void)ws_size;

    const float* x    = (const float*)d_in[0];
    const float* enc  = (const float*)d_in[1];
    const float* relS = (const float*)d_in[2];
    const float* relC = (const float*)d_in[3];
    const float* wq1  = (const float*)d_in[4];
    const float* wk1  = (const float*)d_in[5];
    const float* wv1  = (const float*)d_in[6];
    const float* wq2  = (const float*)d_in[7];
    const float* wk2  = (const float*)d_in[8];
    const float* wv2  = (const float*)d_in[9];
    const float* w1   = (const float*)d_in[10];
    const float* b1   = (const float*)d_in[11];
    const float* w2   = (const float*)d_in[12];
    const float* b2   = (const float*)d_in[13];
    const float* g1   = (const float*)d_in[14];
    const float* be1  = (const float*)d_in[15];
    const float* g2   = (const float*)d_in[16];
    const float* be2  = (const float*)d_in[17];
    const float* g3   = (const float*)d_in[18];
    const float* be3  = (const float*)d_in[19];
    float* out = (float*)d_out;

    // --- workspace bump allocator (256B aligned) ---
    char* ws = (char*)d_ws;
    size_t off = 0;
    auto alloc = [&](size_t bytes) -> char* {
        char* p = ws + off;
        off += (bytes + 255) & ~(size_t)255;
        return p;
    };
    __bf16* wq1b = (__bf16*)alloc((size_t)D_ * D_ * 2);
    __bf16* wk1b = (__bf16*)alloc((size_t)D_ * D_ * 2);
    __bf16* wv1b = (__bf16*)alloc((size_t)D_ * D_ * 2);
    __bf16* wq2b = (__bf16*)alloc((size_t)D_ * D_ * 2);
    __bf16* wk2b = (__bf16*)alloc((size_t)D_ * D_ * 2);
    __bf16* wv2b = (__bf16*)alloc((size_t)D_ * D_ * 2);
    __bf16* w1b  = (__bf16*)alloc((size_t)D_ * FF_ * 2);
    __bf16* w2b  = (__bf16*)alloc((size_t)FF_ * D_ * 2);
    __bf16* xb   = (__bf16*)alloc((size_t)MD_ * D_ * 2);
    __bf16* encb = (__bf16*)alloc((size_t)MD_ * D_ * 2);
    __bf16* tmpb = (__bf16*)alloc((size_t)MD_ * D_ * 2);   // out1/out2 in bf16
    float*  qprj = (float*)alloc((size_t)MD_ * D_ * 4);
    float*  kprj = (float*)alloc((size_t)MD_ * D_ * 4);
    __bf16* vprb = (__bf16*)alloc((size_t)MD_ * D_ * 2);
    float*  Qs   = (float*)alloc((size_t)MD_ * H_ * 4);
    float*  Ks   = (float*)alloc((size_t)MD_ * H_ * 4);
    float*  Asm  = (float*)alloc((size_t)B_ * H_ * S_ * 4);
    __bf16* wsc  = (__bf16*)alloc((size_t)B_ * H_ * S_ * S_ * 2);  // reused: ffh bf16
    float*  attn = (float*)alloc((size_t)MD_ * D_ * 4);
    float*  out1 = (float*)alloc((size_t)MD_ * D_ * 4);
    float*  out2 = (float*)alloc((size_t)MD_ * D_ * 4);
    float*  ffo  = (float*)alloc((size_t)MD_ * D_ * 4);
    __bf16* ffhb = wsc;   // FFN hidden (bf16) reuses scores buffer

    auto cvt = [&](const float* src, __bf16* dst, int n) {
        f32_to_bf16_k<<<(n + 255) / 256, 256, 0, stream>>>(src, dst, n);
    };
    cvt(wq1, wq1b, D_ * D_);  cvt(wk1, wk1b, D_ * D_);  cvt(wv1, wv1b, D_ * D_);
    cvt(wq2, wq2b, D_ * D_);  cvt(wk2, wk2b, D_ * D_);  cvt(wv2, wv2b, D_ * D_);
    cvt(w1,  w1b,  D_ * FF_); cvt(w2,  w2b,  FF_ * D_);
    cvt(x,   xb,   MD_ * D_); cvt(enc, encb, MD_ * D_);

    const dim3 blk(256);
    const dim3 gP(D_ / 128, MD_ / 128, 1);          // projection GEMMs
    const dim3 gAV(1, S_ / 128, B_ * H_);           // batched AV GEMMs
    const long SD = (long)S_ * D_, SS = (long)S_ * S_;

    auto attention = [&](const __bf16* aq, const __bf16* akv, const __bf16* wqb,
                         const __bf16* wkb, const __bf16* wvb, const float* rel,
                         int causal) {
        gemm_bf16_wmma<128,128,0><<<gP, blk, 0, stream>>>(aq,  D_, 0, wqb, D_, 0,0,1, qprj, D_, 0, nullptr, D_);
        gemm_bf16_wmma<128,128,0><<<gP, blk, 0, stream>>>(akv, D_, 0, wkb, D_, 0,0,1, kprj, D_, 0, nullptr, D_);
        gemm_bf16_wmma<128,128,3><<<gP, blk, 0, stream>>>(akv, D_, 0, wvb, D_, 0,0,1, vprb, D_, 0, nullptr, D_);
        chunk_sum64_k<<<(MD_ * H_ * 16) / 256, blk, 0, stream>>>(qprj, Qs, 0.125f); // /sqrt(64)
        chunk_sum64_k<<<(MD_ * H_ * 16) / 256, blk, 0, stream>>>(kprj, Ks, 1.0f);
        skew_colsum_k<<<dim3(2, B_ * H_), blk, 0, stream>>>(rel, Asm);
        softmax_rank1_k<<<dim3(S_, B_ * H_), blk, 0, stream>>>(Qs, Ks, Asm, wsc, causal);
        // AV: per (b,h): [512,512] x [512,64]; V batch offset = b*S*D + h*64
        gemm_bf16_wmma<128,64,0><<<gAV, blk, 0, stream>>>(
            wsc, S_, SS, vprb, D_, SD, 64, H_, attn, 64, (long)S_ * 64, nullptr, S_);
    };

    // ---- self-attention (causal) + LN1 ----
    attention(xb, xb, wq1b, wk1b, wv1b, relS, 1);
    residual_ln_k<<<MD_, blk, 0, stream>>>(x, attn, g1, be1, out1, tmpb);

    // ---- cross-attention + LN2 ----
    attention(tmpb, encb, wq2b, wk2b, wv2b, relC, 0);
    residual_ln_k<<<MD_, blk, 0, stream>>>(out1, attn, g2, be2, out2, tmpb);

    // ---- FFN: relu(out2@w1+b1)@w2+b2, then LN3 -> d_out ----
    gemm_bf16_wmma<128,128,2><<<dim3(FF_ / 128, MD_ / 128, 1), blk, 0, stream>>>(
        tmpb, D_, 0, w1b, FF_, 0, 0, 1, ffhb, FF_, 0, b1, D_);
    gemm_bf16_wmma<128,128,1><<<dim3(D_ / 128, MD_ / 128, 1), blk, 0, stream>>>(
        ffhb, FF_, 0, w2b, D_, 0, 0, 1, ffo, D_, 0, b2, FF_);
    residual_ln_k<<<MD_, blk, 0, stream>>>(ffo, out2, g3, be3, out, nullptr);
}
